// RealMetricsFastSpeech2_24790551233199
// MI455X (gfx1250) — matrix operand order repaired and checked by reading, not compiled, not run on hardware
//
#include <hip/hip_runtime.h>
#include <hip/hip_bf16.h>

// ---------------------------------------------------------------------------
// FastSpeech2 forward, CDNA5 (gfx1250). All big GEMMs (QKV, Wo, conv-FFN via
// implicit im2col, variance-predictor convs, mel projection) run through
// v_wmma_f32_16x16x32_f16 (f16 operands, f32 accumulate). Attention uses a
// flash-attention wave kernel (online softmax). wave32 everywhere.
// ---------------------------------------------------------------------------

typedef __attribute__((ext_vector_type(16))) _Float16 v16h;
typedef __attribute__((ext_vector_type(8)))  float    v8f;

#define BB   16    // batch
#define TT   256   // phoneme length
#define HH   256   // hidden
#define MM   768   // MEL_MAX
#define NHD  2     // heads
#define DHD  128   // head dim
#define NMEL 80

__device__ __forceinline__ float wave_sum(float v) {
#pragma unroll
  for (int o = 16; o > 0; o >>= 1) v += __shfl_xor(v, o, 32);
  return v;
}

// ---------------- weight casting ----------------
__global__ void k_cast_plain(const float* __restrict__ w, _Float16* __restrict__ o, int n) {
  int i = blockIdx.x * blockDim.x + threadIdx.x;
  if (i < n) o[i] = (_Float16)w[i];
}

// conv weight (Cout,Cin,ks) f32 -> (Cout, ks*Cin) f16 with out[co][dk*Cin+ci]=w[co][ci][dk]
__global__ void k_cast_conv(const float* __restrict__ w, _Float16* __restrict__ o,
                            int Cout, int Cin, int ks) {
  int i = blockIdx.x * blockDim.x + threadIdx.x;
  int n = Cout * Cin * ks;
  if (i >= n) return;
  int per = ks * Cin;
  int co = i / per;
  int r  = i - co * per;
  int dk = r / Cin;
  int ci = r - dk * Cin;
  o[i] = (_Float16)w[(co * Cin + ci) * ks + dk];
}

// ---------------- WMMA GEMM: C[M,N] = act(A[M,K] * Bh[N,K]^T + bias) --------
// CONV!=0: implicit im2col, A is X[B*S, Cin] (Cin = 1<<cinShift), k = dk*Cin+ci,
//          row = b*S + t, input sample t+dk-pad (zero padded).
// A-fragment: per lane two contiguous 8-float runs (vectorized, run-uniform
// bounds check in conv mode). B-fragment: 32 contiguous bytes -> v16h load.
template <int CONV>
__device__ __forceinline__ void load_a_run(const float* __restrict__ A, int mrow, int K,
                                           int s, int bb, int t0, int S, int cinShift,
                                           int cmask, int pad, _Float16* dst) {
  const float* p;
  bool ok = true;
  if (CONV) {
    int dk = s >> cinShift;
    int ci = s & cmask;
    int t2 = t0 + dk - pad;
    ok = (t2 >= 0) && (t2 < S);
    p = A + (((size_t)(bb * S + t2)) << cinShift) + ci;
  } else {
    p = A + (size_t)mrow * K + s;
  }
  if (ok) {
    const float4* q = (const float4*)p;
    float4 x0 = q[0];
    float4 x1 = q[1];
    dst[0] = (_Float16)x0.x; dst[1] = (_Float16)x0.y;
    dst[2] = (_Float16)x0.z; dst[3] = (_Float16)x0.w;
    dst[4] = (_Float16)x1.x; dst[5] = (_Float16)x1.y;
    dst[6] = (_Float16)x1.z; dst[7] = (_Float16)x1.w;
  } else {
#pragma unroll
    for (int i = 0; i < 8; ++i) dst[i] = (_Float16)0.f;
  }
}

template <int CONV>
__global__ void k_gemm_wmma(const float* __restrict__ A, const _Float16* __restrict__ Bh,
                            const float* __restrict__ bias, float* __restrict__ C,
                            int M, int N, int K,
                            int S, int cinShift, int pad, int relu) {
  int lane = threadIdx.x & 31;
  int wv   = threadIdx.x >> 5;
  int nt   = blockIdx.x * 4 + wv;
  int mt   = blockIdx.y;
  if (nt * 16 >= N) return;              // wave-uniform exit
  int lg = lane >> 4;                    // lane group (0/1)
  int lr = lane & 15;
  int mrow = mt * 16 + lr;
  int ncol = nt * 16 + lr;
  int bb = 0, t0 = 0, cmask = 0;
  if (CONV) { bb = mrow / S; t0 = mrow - bb * S; cmask = (1 << cinShift) - 1; }
  const _Float16* brow = Bh + (size_t)ncol * K;
  v8f acc = {};
  for (int kk = 0; kk < K; kk += 32) {
    // A fragment: lg0 -> k = kk+{0..15}; lg1 -> kk+{16..31} (contiguous, matches B)
    union { v16h v; _Float16 e[16]; } af;
    int s0 = kk + 16 * lg;
    int s1 = kk + 16 * lg + 8;
    load_a_run<CONV>(A, mrow, K, s0, bb, t0, S, cinShift, cmask, pad, af.e);
    load_a_run<CONV>(A, mrow, K, s1, bb, t0, S, cinShift, cmask, pad, af.e + 8);
    // B fragment: lanes 0-15 -> K=kk..kk+15, lanes 16-31 -> K=kk+16..kk+31 (contiguous)
    v16h bf = *(const v16h*)(brow + kk + 16 * lg);
    acc = __builtin_amdgcn_wmma_f32_16x16x32_f16(false, af.v, false, bf,
                                                 (short)0, acc, false, false);
  }
#pragma unroll
  for (int r = 0; r < 8; ++r) {
    int row = mt * 16 + r + 8 * lg;
    int col = nt * 16 + lr;
    float v = acc[r];
    if (bias) v += bias[col];
    if (relu && v < 0.f) v = 0.f;
    C[(size_t)row * N + col] = v;
  }
}

// ---------------- embedding gather ----------------
__global__ void k_embed(const int* __restrict__ ids, const float* __restrict__ emb,
                        float* __restrict__ out, int n) {
  int i = blockIdx.x * blockDim.x + threadIdx.x;
  if (i >= n) return;
  int bt = i >> 8, h = i & 255;
  out[i] = emb[ids[bt] * HH + h];
}

// ---------------- flash attention (wave per (b,head,q)) ----------------
// qkv: [B*S, 3H]; q at +0, k at +HH, v at +2*HH (per-head offset h*DHD)
__global__ void k_flash_attn(const float* __restrict__ qkv, float* __restrict__ out,
                             const int* __restrict__ klen, int S) {
  int gid  = (blockIdx.x * blockDim.x + threadIdx.x) >> 5;
  int lane = threadIdx.x & 31;
  int b = gid / (NHD * S);
  int r = gid - b * NHD * S;
  int h = r / S;
  int q = r - h * S;
  int d0 = lane * 4;
  const float* qp = qkv + (size_t)(b * S + q) * (3 * HH) + h * DHD;
  float4 qv = *(const float4*)(qp + d0);
  int lim = klen[b];
  float m = -INFINITY, l = 0.f;
  float4 o = {0.f, 0.f, 0.f, 0.f};
  const float scale = 0.088388347648318447f; // 1/sqrt(128)
  for (int j = 0; j < S; ++j) {
    const float* kp = qkv + (size_t)(b * S + j) * (3 * HH) + HH + h * DHD;
    float4 kv = *(const float4*)(kp + d0);
    float p = qv.x * kv.x + qv.y * kv.y + qv.z * kv.z + qv.w * kv.w;
    float s = wave_sum(p) * scale;
    if (j >= lim) s = -1e9f;
    float mn = fmaxf(m, s);
    float c  = __expf(m - mn);
    float p2 = __expf(s - mn);
    const float* vp = qkv + (size_t)(b * S + j) * (3 * HH) + 2 * HH + h * DHD;
    float4 vv = *(const float4*)(vp + d0);
    l = l * c + p2;
    o.x = o.x * c + p2 * vv.x;
    o.y = o.y * c + p2 * vv.y;
    o.z = o.z * c + p2 * vv.z;
    o.w = o.w * c + p2 * vv.w;
    m = mn;
  }
  float inv = 1.f / l;
  float4 res = {o.x * inv, o.y * inv, o.z * inv, o.w * inv};
  float* op = out + (size_t)(b * S + q) * HH + h * DHD;
  *(float4*)(op + d0) = res;
}

// ---------------- layernorm over HH, optional residual add, in-place -------
__global__ void k_ln(float* __restrict__ x, const float* __restrict__ add,
                     const float* __restrict__ g, const float* __restrict__ b, int rows) {
  int wid  = (blockIdx.x * blockDim.x + threadIdx.x) >> 5;
  int lane = threadIdx.x & 31;
  if (wid >= rows) return;
  size_t base = (size_t)wid * HH + lane * 8;
  float v[8], s = 0.f;
  float4 x0 = *(const float4*)(x + base);
  float4 x1 = *(const float4*)(x + base + 4);
  v[0] = x0.x; v[1] = x0.y; v[2] = x0.z; v[3] = x0.w;
  v[4] = x1.x; v[5] = x1.y; v[6] = x1.z; v[7] = x1.w;
  if (add) {
    float4 a0 = *(const float4*)(add + base);
    float4 a1 = *(const float4*)(add + base + 4);
    v[0] += a0.x; v[1] += a0.y; v[2] += a0.z; v[3] += a0.w;
    v[4] += a1.x; v[5] += a1.y; v[6] += a1.z; v[7] += a1.w;
  }
#pragma unroll
  for (int i = 0; i < 8; ++i) s += v[i];
  float mean = wave_sum(s) * (1.f / HH);
  float qv = 0.f;
#pragma unroll
  for (int i = 0; i < 8; ++i) { float d = v[i] - mean; qv += d * d; }
  float rstd = rsqrtf(wave_sum(qv) * (1.f / HH) + 1e-5f);
  float4 g0 = *(const float4*)(g + lane * 8);
  float4 g1 = *(const float4*)(g + lane * 8 + 4);
  float4 b0 = *(const float4*)(b + lane * 8);
  float4 b1 = *(const float4*)(b + lane * 8 + 4);
  float4 r0, r1;
  r0.x = (v[0] - mean) * rstd * g0.x + b0.x;
  r0.y = (v[1] - mean) * rstd * g0.y + b0.y;
  r0.z = (v[2] - mean) * rstd * g0.z + b0.z;
  r0.w = (v[3] - mean) * rstd * g0.w + b0.w;
  r1.x = (v[4] - mean) * rstd * g1.x + b1.x;
  r1.y = (v[5] - mean) * rstd * g1.y + b1.y;
  r1.z = (v[6] - mean) * rstd * g1.z + b1.z;
  r1.w = (v[7] - mean) * rstd * g1.w + b1.w;
  *(float4*)(x + base) = r0;
  *(float4*)(x + base + 4) = r1;
}

// ---------------- variance predictor projection: out[row]=x[row,:]·pw+pb ---
__global__ void k_vp_proj(const float* __restrict__ x, const float* __restrict__ pw,
                          const float* __restrict__ pb, float* __restrict__ out, int rows) {
  int wid  = (blockIdx.x * blockDim.x + threadIdx.x) >> 5;
  int lane = threadIdx.x & 31;
  if (wid >= rows) return;
  size_t base = (size_t)wid * HH + lane * 8;
  float4 x0 = *(const float4*)(x + base);
  float4 x1 = *(const float4*)(x + base + 4);
  float4 w0 = *(const float4*)(pw + lane * 8);
  float4 w1 = *(const float4*)(pw + lane * 8 + 4);
  float s = x0.x * w0.x + x0.y * w0.y + x0.z * w0.z + x0.w * w0.w +
            x1.x * w1.x + x1.y * w1.y + x1.z * w1.z + x1.w * w1.w;
  s = wave_sum(s);
  if (lane == 0) out[wid] = s + pb[0];
}

// ---------------- length regulator ----------------
__global__ void k_lr_scan(const int* __restrict__ dur, int* __restrict__ csum,
                          int* __restrict__ mlen) {
  int b = threadIdx.x;
  if (b >= BB) return;
  int c = 0;
  for (int t = 0; t < TT; ++t) {
    int d = dur[b * TT + t]; if (d < 0) d = 0;
    c += d;
    csum[b * TT + t] = c;
  }
  mlen[b] = c > 1 ? c : 1;
}

__global__ void k_lr_idx(const int* __restrict__ csum, int* __restrict__ idx) {
  int i = blockIdx.x * blockDim.x + threadIdx.x;
  if (i >= BB * MM) return;
  int b = i / MM, pos = i - b * MM;
  const int* c = csum + b * TT;
  int lo = 0, hi = TT;                 // searchsorted(side='right')
  while (lo < hi) { int mid = (lo + hi) >> 1; if (c[mid] <= pos) lo = mid + 1; else hi = mid; }
  idx[i] = lo < TT - 1 ? lo : TT - 1;
}

__global__ void k_gather_dec(const float* __restrict__ xe, const int* __restrict__ idx,
                             const int* __restrict__ mlen, float* __restrict__ dec, int n) {
  int i = blockIdx.x * blockDim.x + threadIdx.x;
  if (i >= n) return;
  int m = i >> 8, h = i & 255;
  int b = m / MM, pos = m - b * MM;
  dec[i] = (pos < mlen[b]) ? xe[((size_t)b * TT + idx[m]) * HH + h] : 0.f;
}

__global__ void k_gather_val(const float* __restrict__ tgt, const int* __restrict__ idx,
                             const int* __restrict__ mlen, float* __restrict__ gv) {
  int i = blockIdx.x * blockDim.x + threadIdx.x;
  if (i >= BB * MM) return;
  int b = i / MM, pos = i - b * MM;
  gv[i] = (pos < mlen[b]) ? tgt[b * TT + idx[i]] : 0.f;
}

// dec[b,m,h] += sum_dk w[h,0,dk]*gv[b,m+dk-1] + bias[h]
__global__ void k_vemb_add(float* __restrict__ dec, const float* __restrict__ gv,
                           const float* __restrict__ w, const float* __restrict__ bias, int n) {
  int i = blockIdx.x * blockDim.x + threadIdx.x;
  if (i >= n) return;
  int m = i >> 8, h = i & 255;
  int b = m / MM, pos = m - b * MM;
  float acc = bias[h];
#pragma unroll
  for (int dk = 0; dk < 3; ++dk) {
    int p = pos + dk - 1;
    if (p >= 0 && p < MM) acc += w[h * 3 + dk] * gv[b * MM + p];
  }
  dec[i] += acc;
}

__global__ void k_write_mlen(const int* __restrict__ mlen, float* __restrict__ out) {
  int i = threadIdx.x;
  if (i < BB) out[i] = (float)mlen[i];
}

// ---------------------------------------------------------------------------
// Host side
// ---------------------------------------------------------------------------
struct FFTP { const float *Wo,*Wqkv,*bo,*bqkv,*c1b,*c1w,*c2b,*c2w,*n1b,*n1g,*n2b,*n2g; };
struct VPP  { const float *c1b,*c1w,*c2b,*c2w,*l1b,*l1g,*l2b,*l2g,*pb,*pw; };
struct VEP  { const float *b,*w; };

static inline void launch_gemm(const float* A, const _Float16* Bh, const float* bias,
                               float* C, int M, int N, int K, int conv, int S,
                               int cinShift, int pad, int relu, hipStream_t st) {
  dim3 g((N / 16 + 3) / 4, M / 16);
  if (conv)
    k_gemm_wmma<1><<<g, 128, 0, st>>>(A, Bh, bias, C, M, N, K, S, cinShift, pad, relu);
  else
    k_gemm_wmma<0><<<g, 128, 0, st>>>(A, Bh, bias, C, M, N, K, S, cinShift, pad, relu);
}
static inline void launch_cast_plain(const float* w, _Float16* o, int n, hipStream_t st) {
  k_cast_plain<<<(n + 255) / 256, 256, 0, st>>>(w, o, n);
}
static inline void launch_cast_conv(const float* w, _Float16* o, int Co, int Ci, int ks,
                                    hipStream_t st) {
  int n = Co * Ci * ks;
  k_cast_conv<<<(n + 255) / 256, 256, 0, st>>>(w, o, Co, Ci, ks);
}

struct Scratch {
  float *x, *qkv, *a, *t, *h, *dec, *gv;
  _Float16* wh;
  int *csum, *idx, *mlen;
};

static void fft_block(float* x, int S, const int* klen, const FFTP& p, Scratch& ws,
                      hipStream_t st) {
  int M = BB * S;
  // QKV projection
  launch_cast_plain(p.Wqkv, ws.wh, 3 * HH * HH, st);
  launch_gemm(x, ws.wh, p.bqkv, ws.qkv, M, 3 * HH, HH, 0, S, 0, 0, 0, st);
  // attention
  k_flash_attn<<<(M * NHD) / 4, 128, 0, st>>>(ws.qkv, ws.a, klen, S);
  // output projection
  launch_cast_plain(p.Wo, ws.wh, HH * HH, st);
  launch_gemm(ws.a, ws.wh, p.bo, ws.t, M, HH, HH, 0, S, 0, 0, 0, st);
  // x = LN(x + attn_proj)
  k_ln<<<M / 8, 256, 0, st>>>(x, ws.t, p.n1g, p.n1b, M);
  // conv1 (K=9, H->4H) + relu, implicit im2col
  launch_cast_conv(p.c1w, ws.wh, 4 * HH, HH, 9, st);
  launch_gemm(x, ws.wh, p.c1b, ws.h, M, 4 * HH, 9 * HH, 1, S, 8, 4, 1, st);
  // conv2 (K=9, 4H->H)
  launch_cast_conv(p.c2w, ws.wh, HH, 4 * HH, 9, st);
  launch_gemm(ws.h, ws.wh, p.c2b, ws.t, M, HH, 9 * 4 * HH, 1, S, 10, 4, 0, st);
  // x = LN(x + ffn)
  k_ln<<<M / 8, 256, 0, st>>>(x, ws.t, p.n2g, p.n2b, M);
}

static void var_pred(const float* x, const VPP& p, float* out, Scratch& ws, hipStream_t st) {
  int M = BB * TT;
  launch_cast_conv(p.c1w, ws.wh, HH, HH, 3, st);
  launch_gemm(x, ws.wh, p.c1b, ws.a, M, HH, 3 * HH, 1, TT, 8, 1, 1, st);
  k_ln<<<M / 8, 256, 0, st>>>(ws.a, nullptr, p.l1g, p.l1b, M);
  launch_cast_conv(p.c2w, ws.wh, HH, HH, 3, st);
  launch_gemm(ws.a, ws.wh, p.c2b, ws.t, M, HH, 3 * HH, 1, TT, 8, 1, 1, st);
  k_ln<<<M / 8, 256, 0, st>>>(ws.t, nullptr, p.l2g, p.l2b, M);
  k_vp_proj<<<M / 8, 256, 0, st>>>(ws.t, p.pw, p.pb, out, M);
}

extern "C" void kernel_launch(void* const* d_in, const int* in_sizes, int n_in,
                              void* d_out, int out_size, void* d_ws, size_t ws_size,
                              hipStream_t stream) {
  (void)in_sizes; (void)n_in; (void)out_size; (void)ws_size;
  // ---- input leaf mapping ----
  // Assumed flattening: top-level dict in insertion order (params first), nested
  // pytrees flattened jax-style (dict keys sorted). params sorted keys:
  // Wm, bm, dec[4], emb, emb_breath, emb_bright, emb_energy, emb_pitch,
  // emb_rough, enc[4], vp_breath, vp_bright, vp_dur, vp_energy, vp_pitch, vp_rough
  int ii = 0;
  auto F = [&]() { return (const float*)d_in[ii++]; };
  const float* Wm = F();
  const float* bm = F();
  FFTP dec4[4], enc4[4];
  auto readFFT = [&](FFTP& p) {
    p.Wo = F(); p.Wqkv = F(); p.bo = F(); p.bqkv = F();
    p.c1b = F(); p.c1w = F(); p.c2b = F(); p.c2w = F();
    p.n1b = F(); p.n1g = F(); p.n2b = F(); p.n2g = F();
  };
  for (int i = 0; i < 4; ++i) readFFT(dec4[i]);
  const float* emb = F();
  VEP e_breath{F(), F()}, e_bright{F(), F()}, e_energy{F(), F()},
      e_pitch{F(), F()}, e_rough{F(), F()};
  for (int i = 0; i < 4; ++i) readFFT(enc4[i]);
  auto readVP = [&](VPP& p) {
    p.c1b = F(); p.c1w = F(); p.c2b = F(); p.c2w = F();
    p.l1b = F(); p.l1g = F(); p.l2b = F(); p.l2g = F();
    p.pb = F(); p.pw = F();
  };
  VPP vp_breath, vp_bright, vp_dur, vp_energy, vp_pitch, vp_rough;
  readVP(vp_breath); readVP(vp_bright); readVP(vp_dur);
  readVP(vp_energy); readVP(vp_pitch); readVP(vp_rough);
  const int* phon    = (const int*)d_in[ii++];
  const int* lengths = (const int*)d_in[ii++];
  const int* tdur    = (const int*)d_in[ii++];
  const float* tp  = F();
  const float* te  = F();
  const float* tb  = F();
  const float* tr  = F();
  const float* tbr = F();

  // ---- workspace layout (floats) ----
  float* wsf = (float*)d_ws;
  Scratch ws;
  size_t off = 0;
  const size_t NROW = (size_t)BB * MM;          // 12288 max rows
  ws.x   = wsf + off; off += NROW * HH;         // 12288*256
  ws.qkv = wsf + off; off += NROW * 3 * HH;     // 12288*768
  ws.a   = wsf + off; off += NROW * HH;
  ws.t   = wsf + off; off += NROW * HH;
  ws.h   = wsf + off; off += NROW * 4 * HH;     // 12288*1024
  ws.dec = wsf + off; off += NROW * HH;
  ws.gv  = wsf + off; off += NROW;              // 16*768
  ws.wh  = (_Float16*)(wsf + off); off += (size_t)(HH) * (9 * 4 * HH) / 2; // 256*9216 halves
  ws.csum = (int*)(wsf + off); off += BB * TT;
  ws.idx  = (int*)(wsf + off); off += BB * MM;
  ws.mlen = (int*)(wsf + off); off += BB;

  // ---- output layout ----
  float* out     = (float*)d_out;
  float* o_mel   = out;                                 // [B, MM, NMEL]
  float* o_ldur  = out + (size_t)BB * MM * NMEL;        // [B, T]
  float* o_pitch = o_ldur + BB * TT;
  float* o_energy= o_pitch + BB * TT;
  float* o_breath= o_energy + BB * TT;
  float* o_rough = o_breath + BB * TT;
  float* o_bright= o_rough + BB * TT;
  float* o_mlen  = o_bright + BB * TT;

  // ---- encoder ----
  {
    int n = BB * TT * HH;
    k_embed<<<(n + 255) / 256, 256, 0, stream>>>(phon, emb, ws.x, n);
  }
  for (int i = 0; i < 4; ++i) fft_block(ws.x, TT, lengths, enc4[i], ws, stream);

  // ---- variance predictors (on encoder output) ----
  var_pred(ws.x, vp_dur,    o_ldur,   ws, stream);
  var_pred(ws.x, vp_pitch,  o_pitch,  ws, stream);
  var_pred(ws.x, vp_energy, o_energy, ws, stream);
  var_pred(ws.x, vp_breath, o_breath, ws, stream);
  var_pred(ws.x, vp_rough,  o_rough,  ws, stream);
  var_pred(ws.x, vp_bright, o_bright, ws, stream);

  // ---- length regulator ----
  k_lr_scan<<<1, 32, 0, stream>>>(tdur, ws.csum, ws.mlen);
  k_lr_idx<<<(BB * MM + 255) / 256, 256, 0, stream>>>(ws.csum, ws.idx);
  {
    int n = BB * MM * HH;
    k_gather_dec<<<(n + 255) / 256, 256, 0, stream>>>(ws.x, ws.idx, ws.mlen, ws.dec, n);
    // variance embeddings (pitch, energy, breath, rough, bright)
    const float* tgts[5] = {tp, te, tb, tr, tbr};
    const VEP*   embs[5] = {&e_pitch, &e_energy, &e_breath, &e_rough, &e_bright};
    for (int v = 0; v < 5; ++v) {
      k_gather_val<<<(BB * MM + 255) / 256, 256, 0, stream>>>(tgts[v], ws.idx, ws.mlen, ws.gv);
      k_vemb_add<<<(n + 255) / 256, 256, 0, stream>>>(ws.dec, ws.gv, embs[v]->w, embs[v]->b, n);
    }
  }

  // ---- decoder ----
  for (int i = 0; i < 4; ++i) fft_block(ws.dec, MM, ws.mlen, dec4[i], ws, stream);

  // ---- mel projection [12288,256]x[80,256]^T ----
  launch_cast_plain(Wm, ws.wh, NMEL * HH, stream);
  launch_gemm(ws.dec, ws.wh, bm, o_mel, BB * MM, NMEL, HH, 0, MM, 0, 0, 0, stream);

  // ---- mel_len ----
  k_write_mlen<<<1, 32, 0, stream>>>(ws.mlen, o_mlen);
}
